// Decoder_3461743640809
// MI455X (gfx1250) — compile-verified, matrix operand order checked by reading, not run on hardware
//
#include <hip/hip_runtime.h>
#include <hip/hip_bf16.h>

// ---------------------------------------------------------------------------
// Tacotron2-style decoder for MI455X (gfx1250, wave32).
//  * One-time prep: f16 weight concat (w_ih|w_hh), prenet for all 500 steps,
//    pm = memory @ wm.T. Recurrent weights (~36MB f16) stay L2-resident.
//  * Per step (500 sequential iterations, 4 kernels each):
//      1) gemm_gates<8,16,32>  : attention-LSTM gates via v_wmma f16->f32
//      2) attn_step            : LSTM pointwise + location attention + softmax
//      3) gemm_gates<32,16,32> : decoder-LSTM gates via v_wmma
//      4) proj_step            : LSTM pointwise + mel/gate projection
//  * GEMM K-loop is split into 3 compile-time-sized segments (one per A
//    source) with double-buffered fragments so loads pipeline past the WMMA.
// ---------------------------------------------------------------------------

typedef __attribute__((ext_vector_type(16))) _Float16 v16h;
typedef __attribute__((ext_vector_type(8)))  _Float16 v8h;
typedef __attribute__((ext_vector_type(8)))  float    v8f;

#define B_SZ    32
#define T_IN    512
#define T_OUT   500
#define N_MEL   80
#define PRENET  256
#define RNN     1024
#define ATT_DIM 128
#define LOC_F   32
#define LOC_K   31
#define E_DIM   512
#define NGATE   4096

__device__ __forceinline__ float sigf(float x) { return 1.0f / (1.0f + __expf(-x)); }

// 16x32 f16 fragment load (A layout; B reads rows of row-major [N][K] weights
// with N on lanes). CDNA5 16-bit layout: lane l -> row r=l&15, h=l>>4;
// element j: K = (j&7) + 8*h + 16*(j>>3) -> two contiguous 8-half runs.
__device__ __forceinline__ v16h load_frag(const _Float16* __restrict__ base,
                                          int stride, int row, int k0, int h) {
    const _Float16* p = base + (size_t)row * stride + k0 + h * 8;
    v8h lo = *(const v8h*)(p);
    v8h hi = *(const v8h*)(p + 16);
    return __builtin_shufflevector(lo, hi, 0, 1, 2, 3, 4, 5, 6, 7,
                                            8, 9, 10, 11, 12, 13, 14, 15);
}

// One K segment of the gate GEMM: NCHUNK compile-time 32-wide K chunks from a
// single A source. Double-buffered fragments -> loads overlap the WMMA.
template <int NCHUNK>
__device__ __forceinline__ void mma_seg(const _Float16* __restrict__ sp, int ss,
                                        const _Float16* __restrict__ Wseg, int Ktot,
                                        int arow, int ncol, int h, v8f& c) {
    v16h a = load_frag(sp, ss, arow, 0, h);
    v16h b = load_frag(Wseg, Ktot, ncol, 0, h);
#pragma unroll
    for (int i = 1; i < NCHUNK; ++i) {
        v16h an = load_frag(sp, ss, arow, i * 32, h);
        v16h bn = load_frag(Wseg, Ktot, ncol, i * 32, h);
        __builtin_prefetch(Wseg + (size_t)ncol * Ktot + i * 32 + 32, 0, 3);
        c = __builtin_amdgcn_wmma_f32_16x16x32_f16(
                false, a, false, b, (short)0, c, false, false);
        a = an; b = bn;
    }
    c = __builtin_amdgcn_wmma_f32_16x16x32_f16(
            false, a, false, b, (short)0, c, false, false);
}

// ---------------------------------------------------------------------------
// LSTM gate GEMM: gates[32][4096] = A[32][Ktot] @ W[Ktot][4096]
// W row-major [4096][Ktot] f16 (B(k,n) = W[n][k], contiguous in k).
// A = [seg0 | seg1 | seg2], Ktot = 32*(N0+N1+N2).
// Grid: 128 blocks x 128 threads (4 waves). One wave per 16x16 output tile.
// ---------------------------------------------------------------------------
template <int N0, int N1, int N2>
__global__ void gemm_gates(const _Float16* __restrict__ W,
                           const _Float16* __restrict__ s0, int s0s,
                           const _Float16* __restrict__ s1, int s1s,
                           const _Float16* __restrict__ s2, int s2s,
                           float* __restrict__ gates) {
    constexpr int Ktot = 32 * (N0 + N1 + N2);
    const int lane  = threadIdx.x & 31;
    const int wave  = threadIdx.x >> 5;
    const int gw    = blockIdx.x * 4 + wave;   // 0..511
    const int mtile = gw & 1;                  // batch rows 0-15 / 16-31
    const int ntile = gw >> 1;                 // 0..255
    const int h     = lane >> 4;
    const int r     = lane & 15;
    const int ncol  = ntile * 16 + r;
    const int arow  = mtile * 16 + r;

    v8f c = {};
    mma_seg<N0>(s0, s0s, W,                   Ktot, arow, ncol, h, c);
    mma_seg<N1>(s1, s1s, W + N0 * 32,         Ktot, arow, ncol, h, c);
    mma_seg<N2>(s2, s2s, W + (N0 + N1) * 32,  Ktot, arow, ncol, h, c);

#pragma unroll
    for (int i = 0; i < 8; ++i) {
        gates[(size_t)(mtile * 16 + i + 8 * h) * NGATE + ncol] = c[i];
    }
}

// ---------------------------------------------------------------------------
// Attention step: att-LSTM pointwise, pq, location conv+dense, energies,
// softmax, context. One block (512 threads) per batch element.
// ---------------------------------------------------------------------------
#define ANT 512   // attention kernel block size
__global__ void attn_step(const float* __restrict__ gates,
                          const float* __restrict__ b_ih, const float* __restrict__ b_hh,
                          float* __restrict__ ac,
                          float* __restrict__ aw, float* __restrict__ awc,
                          const float* __restrict__ pm,
                          const float* __restrict__ memory,
                          const float* __restrict__ wq,
                          const float* __restrict__ vvec,
                          const float* __restrict__ loc_conv_w,
                          const float* __restrict__ loc_dense_w,
                          float* __restrict__ actx32, _Float16* __restrict__ actx16,
                          _Float16* __restrict__ ah16,
                          float* __restrict__ out_align, int t) {
    const int b   = blockIdx.x;
    const int tid = threadIdx.x;
    __shared__ float sm_ah[RNN];
    __shared__ float sm_aw[T_IN];
    __shared__ float sm_awc[T_IN];
    __shared__ float sm_pq[ATT_DIM];
    __shared__ float sm_e[T_IN];
    __shared__ float sm_red[ANT];

    // 1) attention LSTM pointwise (gate order i,f,g,o)
    for (int n = tid; n < RNN; n += ANT) {
        const size_t gbase = (size_t)b * NGATE + n;
        const float gi = gates[gbase]          + b_ih[n]          + b_hh[n];
        const float gf = gates[gbase + 1024]   + b_ih[n + 1024]   + b_hh[n + 1024];
        const float gg = gates[gbase + 2048]   + b_ih[n + 2048]   + b_hh[n + 2048];
        const float go = gates[gbase + 3072]   + b_ih[n + 3072]   + b_hh[n + 3072];
        const float c_old = ac[b * RNN + n];
        const float c2 = sigf(gf) * c_old + sigf(gi) * tanhf(gg);
        const float h2 = sigf(go) * tanhf(c2);
        ac[b * RNN + n]   = c2;
        sm_ah[n]          = h2;
        ah16[b * RNN + n] = (_Float16)h2;
    }
    {
        const int i = tid;                    // ANT == T_IN: one element each
        sm_aw[i]  = aw[b * T_IN + i];
        sm_awc[i] = awc[b * T_IN + i];
    }
    __syncthreads();

    // 2) pq = ah @ wq.T   [128]
    if (tid < ATT_DIM) {
        float acc = 0.0f;
        const float* w = wq + tid * RNN;
        for (int k = 0; k < RNN; ++k) acc += sm_ah[k] * w[k];
        sm_pq[tid] = acc;
    }
    __syncthreads();

    // 3) location conv (pad 15) + dense + energies; one time index per thread
    {
        const int ti = tid;
        float lf[LOC_F];
#pragma unroll
        for (int f = 0; f < LOC_F; ++f) lf[f] = 0.0f;
        for (int k = 0; k < LOC_K; ++k) {
            const int src = ti + k - 15;
            const bool in = (src >= 0) && (src < T_IN);
            const float a0 = in ? sm_aw[src]  : 0.0f;
            const float a1 = in ? sm_awc[src] : 0.0f;
#pragma unroll
            for (int f = 0; f < LOC_F; ++f) {
                lf[f] += loc_conv_w[f * 62 + k]         * a0
                       + loc_conv_w[f * 62 + LOC_K + k] * a1;
            }
        }
        float e = 0.0f;
        const float* pmr = pm + ((size_t)(b * T_IN + ti)) * ATT_DIM;
        for (int a = 0; a < ATT_DIM; ++a) {
            float s = sm_pq[a] + pmr[a];
            const float* ld = loc_dense_w + a * LOC_F;
#pragma unroll
            for (int f = 0; f < LOC_F; ++f) s += lf[f] * ld[f];
            e += vvec[a] * tanhf(s);
        }
        sm_e[ti] = e;
    }
    __syncthreads();

    // 4) softmax over T_IN
    sm_red[tid] = sm_e[tid];
    __syncthreads();
    for (int s = ANT / 2; s > 0; s >>= 1) {
        if (tid < s) sm_red[tid] = fmaxf(sm_red[tid], sm_red[tid + s]);
        __syncthreads();
    }
    const float mx = sm_red[0];
    __syncthreads();
    const float ev = __expf(sm_e[tid] - mx);
    sm_e[tid] = ev;
    sm_red[tid] = ev;
    __syncthreads();
    for (int s = ANT / 2; s > 0; s >>= 1) {
        if (tid < s) sm_red[tid] += sm_red[tid + s];
        __syncthreads();
    }
    const float inv = 1.0f / sm_red[0];
    __syncthreads();
    {
        const int ti = tid;
        const float w2 = sm_e[ti] * inv;
        sm_e[ti]           = w2;              // reuse as aw2
        aw[b * T_IN + ti]  = w2;
        awc[b * T_IN + ti] = sm_awc[ti] + w2;
        out_align[((size_t)b * T_OUT + t) * T_IN + ti] = w2;
    }
    __syncthreads();

    // 5) context = aw2 @ memory[b]   [512]; one embedding column per thread
    {
        const int e0 = tid;
        float acc = 0.0f;
        const float* mb = memory + (size_t)b * T_IN * E_DIM + e0;
        for (int ti = 0; ti < T_IN; ++ti) acc += sm_e[ti] * mb[(size_t)ti * E_DIM];
        actx32[b * E_DIM + e0] = acc;
        actx16[b * E_DIM + e0] = (_Float16)acc;
    }
}

// ---------------------------------------------------------------------------
// Decoder-LSTM pointwise + mel/gate projection. One block per batch element.
// ---------------------------------------------------------------------------
__global__ void proj_step(const float* __restrict__ gates,
                          const float* __restrict__ b_ih, const float* __restrict__ b_hh,
                          float* __restrict__ dc,
                          const float* __restrict__ actx32,
                          _Float16* __restrict__ dh16,
                          const float* __restrict__ proj_w, const float* __restrict__ proj_b,
                          const float* __restrict__ gate_w, const float* __restrict__ gate_b,
                          float* __restrict__ out_mel, float* __restrict__ out_gate, int t) {
    const int b   = blockIdx.x;
    const int tid = threadIdx.x;
    __shared__ float sm_dh[RNN];
    __shared__ float sm_cx[E_DIM];

    for (int n = tid; n < RNN; n += 256) {
        const size_t gbase = (size_t)b * NGATE + n;
        const float gi = gates[gbase]          + b_ih[n]          + b_hh[n];
        const float gf = gates[gbase + 1024]   + b_ih[n + 1024]   + b_hh[n + 1024];
        const float gg = gates[gbase + 2048]   + b_ih[n + 2048]   + b_hh[n + 2048];
        const float go = gates[gbase + 3072]   + b_ih[n + 3072]   + b_hh[n + 3072];
        const float c_old = dc[b * RNN + n];
        const float c2 = sigf(gf) * c_old + sigf(gi) * tanhf(gg);
        const float h2 = sigf(go) * tanhf(c2);
        dc[b * RNN + n]   = c2;
        sm_dh[n]          = h2;
        dh16[b * RNN + n] = (_Float16)h2;
    }
    for (int i = tid; i < E_DIM; i += 256) sm_cx[i] = actx32[b * E_DIM + i];
    __syncthreads();

    for (int o = tid; o < N_MEL + 1; o += 256) {
        const float* w = (o < N_MEL) ? (proj_w + (size_t)o * 1536) : gate_w;
        float acc = (o < N_MEL) ? proj_b[o] : gate_b[0];
        for (int k = 0; k < RNN; ++k)   acc += w[k] * sm_dh[k];
        for (int k = 0; k < E_DIM; ++k) acc += w[RNN + k] * sm_cx[k];
        if (o < N_MEL) out_mel[((size_t)b * N_MEL + o) * T_OUT + t] = acc;
        else           out_gate[b * T_OUT + t] = acc;
    }
}

// ---------------------------------------------------------------------------
// One-time prep kernels
// ---------------------------------------------------------------------------
__global__ void convert_cat(const float* __restrict__ A, int Ka,
                            const float* __restrict__ Bm, int Kb,
                            _Float16* __restrict__ out, long total) {
    const long Kt = Ka + Kb;
    for (long i = (long)blockIdx.x * blockDim.x + threadIdx.x; i < total;
         i += (long)gridDim.x * blockDim.x) {
        const long n = i / Kt, k = i % Kt;
        const float val = (k < Ka) ? A[n * Ka + k] : Bm[n * Kb + (k - Ka)];
        out[i] = (_Float16)val;
    }
}

__global__ void prenet_kernel(const float* __restrict__ mels,
                              const float* __restrict__ w1,
                              const float* __restrict__ w2,
                              _Float16* __restrict__ pre) {
    const int blk = blockIdx.x;   // t*32 + b
    const int t   = blk >> 5;
    const int b   = blk & 31;
    const int tid = threadIdx.x;
    __shared__ float sm_x[N_MEL];
    __shared__ float sm_h1[PRENET];
    if (tid < N_MEL)
        sm_x[tid] = (t == 0) ? 0.0f
                             : mels[((size_t)b * N_MEL + tid) * T_OUT + (t - 1)];
    __syncthreads();
    {
        float acc = 0.0f;
        const float* w = w1 + tid * N_MEL;
        for (int m = 0; m < N_MEL; ++m) acc += sm_x[m] * w[m];
        sm_h1[tid] = fmaxf(acc, 0.0f);
    }
    __syncthreads();
    {
        float acc = 0.0f;
        const float* w = w2 + tid * PRENET;
        for (int k = 0; k < PRENET; ++k) acc += sm_h1[k] * w[k];
        pre[(size_t)blk * PRENET + tid] = (_Float16)fmaxf(acc, 0.0f);
    }
}

__global__ void pm_kernel(const float* __restrict__ memory,
                          const float* __restrict__ wm,
                          float* __restrict__ pm) {
    const int row = blockIdx.x;          // b*512 + t
    const int a   = threadIdx.x;         // 0..127
    const float* mr = memory + (size_t)row * E_DIM;
    const float* w  = wm + (size_t)a * E_DIM;
    float acc = 0.0f;
    for (int k = 0; k < E_DIM; ++k) acc += mr[k] * w[k];
    pm[(size_t)row * ATT_DIM + a] = acc;
}

__global__ void zero_f32(float* p, int n) {
    int i = blockIdx.x * blockDim.x + threadIdx.x;
    if (i < n) p[i] = 0.0f;
}
__global__ void zero_f16(_Float16* p, int n) {
    int i = blockIdx.x * blockDim.x + threadIdx.x;
    if (i < n) p[i] = (_Float16)0.0f;
}

// ---------------------------------------------------------------------------
extern "C" void kernel_launch(void* const* d_in, const int* in_sizes, int n_in,
                              void* d_out, int out_size, void* d_ws, size_t ws_size,
                              hipStream_t stream) {
    (void)in_sizes; (void)n_in; (void)out_size; (void)ws_size;
    const float* memory     = (const float*)d_in[0];
    const float* mels       = (const float*)d_in[1];
    const float* prenet_w1  = (const float*)d_in[2];
    const float* prenet_w2  = (const float*)d_in[3];
    const float* att_w_ih   = (const float*)d_in[4];
    const float* att_w_hh   = (const float*)d_in[5];
    const float* att_b_ih   = (const float*)d_in[6];
    const float* att_b_hh   = (const float*)d_in[7];
    const float* wq         = (const float*)d_in[8];
    const float* wm         = (const float*)d_in[9];
    const float* v          = (const float*)d_in[10];
    const float* loc_conv_w = (const float*)d_in[11];
    const float* loc_dense_w= (const float*)d_in[12];
    const float* dec_w_ih   = (const float*)d_in[13];
    const float* dec_w_hh   = (const float*)d_in[14];
    const float* dec_b_ih   = (const float*)d_in[15];
    const float* dec_b_hh   = (const float*)d_in[16];
    const float* proj_w     = (const float*)d_in[17];
    const float* proj_b     = (const float*)d_in[18];
    const float* gate_w     = (const float*)d_in[19];
    const float* gate_b     = (const float*)d_in[20];

    float* out_mel   = (float*)d_out;
    float* out_gate  = out_mel + (size_t)B_SZ * N_MEL * T_OUT;
    float* out_align = out_gate + (size_t)B_SZ * T_OUT;

    // bump allocator over workspace (total ~54 MB)
    char* ws = (char*)d_ws;
    size_t off = 0;
    auto alloc = [&](size_t bytes) -> void* {
        void* p = ws + off;
        off = (off + bytes + 255) & ~(size_t)255;
        return p;
    };
    _Float16* Wcat_att = (_Float16*)alloc((size_t)NGATE * 1792 * 2);
    _Float16* Wcat_dec = (_Float16*)alloc((size_t)NGATE * 2560 * 2);
    _Float16* pre      = (_Float16*)alloc((size_t)T_OUT * B_SZ * PRENET * 2);
    float*    pm       = (float*)   alloc((size_t)B_SZ * T_IN * ATT_DIM * 4);
    float*    gates_a  = (float*)   alloc((size_t)B_SZ * NGATE * 4);
    float*    gates_d  = (float*)   alloc((size_t)B_SZ * NGATE * 4);
    float*    ac       = (float*)   alloc((size_t)B_SZ * RNN * 4);
    float*    dc       = (float*)   alloc((size_t)B_SZ * RNN * 4);
    float*    aw       = (float*)   alloc((size_t)B_SZ * T_IN * 4);
    float*    awc      = (float*)   alloc((size_t)B_SZ * T_IN * 4);
    float*    actx32   = (float*)   alloc((size_t)B_SZ * E_DIM * 4);
    _Float16* actx16   = (_Float16*)alloc((size_t)B_SZ * E_DIM * 2);
    _Float16* ah16     = (_Float16*)alloc((size_t)B_SZ * RNN * 2);
    _Float16* dh16     = (_Float16*)alloc((size_t)B_SZ * RNN * 2);

    // ---- one-time prep ----
    convert_cat<<<2048, 256, 0, stream>>>(att_w_ih, PRENET + E_DIM, att_w_hh, RNN,
                                          Wcat_att, (long)NGATE * 1792);
    convert_cat<<<2048, 256, 0, stream>>>(dec_w_ih, RNN + E_DIM, dec_w_hh, RNN,
                                          Wcat_dec, (long)NGATE * 2560);
    prenet_kernel<<<T_OUT * B_SZ, 256, 0, stream>>>(mels, prenet_w1, prenet_w2, pre);
    pm_kernel<<<B_SZ * T_IN, 128, 0, stream>>>(memory, wm, pm);
    zero_f32<<<(B_SZ * RNN + 255) / 256, 256, 0, stream>>>(ac, B_SZ * RNN);
    zero_f32<<<(B_SZ * RNN + 255) / 256, 256, 0, stream>>>(dc, B_SZ * RNN);
    zero_f32<<<(B_SZ * T_IN + 255) / 256, 256, 0, stream>>>(aw, B_SZ * T_IN);
    zero_f32<<<(B_SZ * T_IN + 255) / 256, 256, 0, stream>>>(awc, B_SZ * T_IN);
    zero_f16<<<(B_SZ * E_DIM + 255) / 256, 256, 0, stream>>>(actx16, B_SZ * E_DIM);
    zero_f16<<<(B_SZ * RNN + 255) / 256, 256, 0, stream>>>(ah16, B_SZ * RNN);
    zero_f16<<<(B_SZ * RNN + 255) / 256, 256, 0, stream>>>(dh16, B_SZ * RNN);

    // ---- sequential decode loop (graph-captured: 2000 launches) ----
    for (int t = 0; t < T_OUT; ++t) {
        // attention LSTM gates: A = [pre_t(8x32) | actx(16x32) | ah(32x32)]
        gemm_gates<8, 16, 32><<<128, 128, 0, stream>>>(
            Wcat_att,
            pre + (size_t)t * B_SZ * PRENET, PRENET,
            actx16, E_DIM,
            ah16, RNN,
            gates_a);
        attn_step<<<B_SZ, ANT, 0, stream>>>(
            gates_a, att_b_ih, att_b_hh, ac, aw, awc, pm, memory, wq, v,
            loc_conv_w, loc_dense_w, actx32, actx16, ah16, out_align, t);
        // decoder LSTM gates: A = [ah(32x32) | actx(16x32) | dh(32x32)]
        gemm_gates<32, 16, 32><<<128, 128, 0, stream>>>(
            Wcat_dec,
            ah16, RNN,
            actx16, E_DIM,
            dh16, RNN,
            gates_d);
        proj_step<<<B_SZ, 256, 0, stream>>>(
            gates_d, dec_b_ih, dec_b_hh, dc, actx32, dh16,
            proj_w, proj_b, gate_w, gate_b, out_mel, out_gate, t);
    }
}
